// TopKTokenChoiceRouter_65481071411007
// MI455X (gfx1250) — matrix operand order repaired and checked by reading, not compiled, not run on hardware
//
#include <hip/hip_runtime.h>
#include <hip/hip_bf16.h>

// ---------------------------------------------------------------------------
// MoE Top-K token-choice router, MI455X (gfx1250, wave32).
//   tokens : [16384, 4096] f32 (x reshaped)
//   W      : [128, 4096]   f32 (gating weights, row-major)
//   out    : [16384*8] f32 weights ++ [16384*8] f32 (indices as float)
// Forward aux-loss term is identically zero -> GEMM + softmax + top-8 only.
//
// Roofline: x = 256 MB streamed once (11.5 us floor @ 23.3 TB/s). GEMM runs
// as bf16 WMMA (v_wmma_f32_16x16x32_bf16) so compute sits far below that
// floor. W is pre-converted to bf16 in d_ws so the B path is pure b128 loads.
// 64-token blocks (256 blocks x 8 waves = 2048 waves) keep every SIMD fed;
// each wave owns 2 M-tiles x 2 N-tiles so both A and B fragments feed two
// WMMAs each. LDS = 32 KB/block -> many co-resident blocks per 320 KB WGP.
// ---------------------------------------------------------------------------

#define NUM_TOKENS 16384
#define HID        4096
#define NEXP       128
#define TOPK       8
#define TOK_BLK    64

typedef __attribute__((ext_vector_type(16))) __bf16 v16bf;
typedef __attribute__((ext_vector_type(8)))  __bf16 v8bf;
typedef __attribute__((ext_vector_type(8)))  float  v8f;
typedef __attribute__((ext_vector_type(4)))  float  v4f;

__device__ __forceinline__ v16bf pack16_bf16(v4f f0, v4f f1, v4f f2, v4f f3) {
  v16bf r;
  r[0]  = (__bf16)f0[0]; r[1]  = (__bf16)f0[1]; r[2]  = (__bf16)f0[2]; r[3]  = (__bf16)f0[3];
  r[4]  = (__bf16)f1[0]; r[5]  = (__bf16)f1[1]; r[6]  = (__bf16)f1[2]; r[7]  = (__bf16)f1[3];
  r[8]  = (__bf16)f2[0]; r[9]  = (__bf16)f2[1]; r[10] = (__bf16)f2[2]; r[11] = (__bf16)f2[3];
  r[12] = (__bf16)f3[0]; r[13] = (__bf16)f3[1]; r[14] = (__bf16)f3[2]; r[15] = (__bf16)f3[3];
  return r;
}

// A fragment (16x32 bf16): lane (m = lane&15, khalf = lane>>4) holds
// K = khalf*16 + {0..15}. x has no reuse -> NT loads.
__device__ __forceinline__ v16bf load_a_frag(const float* __restrict__ aRow,
                                             int kb, int khalf) {
  const float* pa = aRow + kb + khalf * 16;
  v4f a0 = __builtin_nontemporal_load((const v4f*)(pa));
  v4f a1 = __builtin_nontemporal_load((const v4f*)(pa + 4));
  v4f a2 = __builtin_nontemporal_load((const v4f*)(pa + 8));
  v4f a3 = __builtin_nontemporal_load((const v4f*)(pa + 12));
  return pack16_bf16(a0, a1, a2, a3);
}

// --- W f32 -> bf16 pre-conversion (128*4096 elems, 8 per thread) ------------
__global__ __launch_bounds__(256) void w_to_bf16_kernel(
    const float* __restrict__ Wf, __bf16* __restrict__ Wb) {
  const int i = (blockIdx.x * 256 + threadIdx.x) * 8;
  v4f f0 = *(const v4f*)(Wf + i);
  v4f f1 = *(const v4f*)(Wf + i + 4);
  v8bf r;
  r[0] = (__bf16)f0[0]; r[1] = (__bf16)f0[1]; r[2] = (__bf16)f0[2]; r[3] = (__bf16)f0[3];
  r[4] = (__bf16)f1[0]; r[5] = (__bf16)f1[1]; r[6] = (__bf16)f1[2]; r[7] = (__bf16)f1[3];
  *(v8bf*)(Wb + i) = r;
}

// --- Main fused kernel ------------------------------------------------------
template <bool PRECONV>
__global__ __launch_bounds__(256) void moe_router_kernel(
    const float* __restrict__ x, const float* __restrict__ Wf,
    const __bf16* __restrict__ Wb, float* __restrict__ out) {
  __shared__ float sLogits[TOK_BLK * NEXP];  // 32 KB: 64 tokens x 128 experts

  const int lane    = threadIdx.x & 31;
  const int wave    = threadIdx.x >> 5;
  const int n       = lane & 15;  // column within a 16-wide tile
  const int khalf   = lane >> 4;  // which K-half this lane holds
  const int tokBase = blockIdx.x * TOK_BLK;

  // Wave decomposition: 2 M-pairs x 4 N-quads.
  const int mPair = wave & 1;   // tokens [tokBase + mPair*32, +32)
  const int nQuad = wave >> 1;  // experts [nQuad*32, +32) = N-tiles {2q, 2q+1}

  // ------------------- Phase 1: bf16 WMMA gating GEMM -------------------
  const float* aRow0 = x + (size_t)(tokBase + mPair * 32 + n) * HID;
  const float* aRow1 = aRow0 + (size_t)16 * HID;

  v8f acc[2][2] = {};  // [mt][nt]

#pragma unroll 2
  for (int kb = 0; kb < HID; kb += 32) {
    v16bf a0 = load_a_frag(aRow0, kb, khalf);
    v16bf a1 = load_a_frag(aRow1, kb, khalf);

#pragma unroll
    for (int nt = 0; nt < 2; ++nt) {
      // B (32x16 bf16): lane holds column e, K = khalf*16 + {0..15}
      const int e = (nQuad * 2 + nt) * 16 + n;
      v16bf bfrag;
      if constexpr (PRECONV) {
        bfrag = *(const v16bf*)(Wb + (size_t)e * HID + kb + khalf * 16);  // 2x b128, no VALU
      } else {
        const float* pb = Wf + (size_t)e * HID + kb + khalf * 16;
        bfrag = pack16_bf16(*(const v4f*)(pb), *(const v4f*)(pb + 4),
                            *(const v4f*)(pb + 8), *(const v4f*)(pb + 12));
      }
      acc[0][nt] = __builtin_amdgcn_wmma_f32_16x16x32_bf16(
          false, a0, false, bfrag, (short)0, acc[0][nt], false, false);
      acc[1][nt] = __builtin_amdgcn_wmma_f32_16x16x32_bf16(
          false, a1, false, bfrag, (short)0, acc[1][nt], false, false);
    }
  }

  // Park logits in LDS: C/D layout -> row M = v + 8*khalf, col N = lane&15
#pragma unroll
  for (int mt = 0; mt < 2; ++mt) {
#pragma unroll
    for (int nt = 0; nt < 2; ++nt) {
#pragma unroll
      for (int v = 0; v < 8; ++v) {
        const int rowLocal = mPair * 32 + mt * 16 + v + 8 * khalf;
        const int col      = (nQuad * 2 + nt) * 16 + n;
        sLogits[rowLocal * NEXP + col] = acc[mt][nt][v];
      }
    }
  }
  __syncthreads();

  // ------------------- Phase 2: softmax + top-8 per token -------------------
  // Wave w handles 8 tokens; lane holds experts {lane, +32, +64, +96}.
  const int wBase = NUM_TOKENS * TOPK;  // offset of index block in out
  for (int tl = 0; tl < TOK_BLK / 8; ++tl) {
    const int rowLocal = wave * (TOK_BLK / 8) + tl;
    const int t        = tokBase + rowLocal;
    const float* lr    = &sLogits[rowLocal * NEXP];

    float l0 = lr[lane];
    float l1 = lr[lane + 32];
    float l2 = lr[lane + 64];
    float l3 = lr[lane + 96];

    float m = fmaxf(fmaxf(l0, l1), fmaxf(l2, l3));
#pragma unroll
    for (int off = 16; off > 0; off >>= 1)
      m = fmaxf(m, __shfl_xor(m, off, 32));

    float p[4];
    p[0] = __expf(l0 - m);
    p[1] = __expf(l1 - m);
    p[2] = __expf(l2 - m);
    p[3] = __expf(l3 - m);

    float s = p[0] + p[1] + p[2] + p[3];
#pragma unroll
    for (int off = 16; off > 0; off >>= 1)
      s += __shfl_xor(s, off, 32);
    const float inv = __frcp_rn(s);

    // iterative wave-argmax top-8 (ties -> lower expert index, like lax.top_k)
    int takenMask = 0;
#pragma unroll
    for (int k = 0; k < TOPK; ++k) {
      float bv = -1.0f;  // probs >= 0, so -1 marks "taken"
      int   bi = 0x7fffffff;
#pragma unroll
      for (int j = 0; j < 4; ++j) {
        float vj = ((takenMask >> j) & 1) ? -1.0f : p[j];
        int   ij = j * 32 + lane;
        if (vj > bv || (vj == bv && ij < bi)) { bv = vj; bi = ij; }
      }
#pragma unroll
      for (int off = 16; off > 0; off >>= 1) {
        float ov = __shfl_xor(bv, off, 32);
        int   oi = __shfl_xor(bi, off, 32);
        if (ov > bv || (ov == bv && oi < bi)) { bv = ov; bi = oi; }
      }
      if ((bi & 31) == lane) takenMask |= 1 << (bi >> 5);
      if (lane == 0) {
        out[t * TOPK + k]         = bv * inv;   // expert weight
        out[wBase + t * TOPK + k] = (float)bi;  // expert index
      }
    }
  }
}

extern "C" void kernel_launch(void* const* d_in, const int* in_sizes, int n_in,
                              void* d_out, int out_size, void* d_ws, size_t ws_size,
                              hipStream_t stream) {
  (void)in_sizes; (void)n_in; (void)out_size;
  const float* x = (const float*)d_in[0];  // [16384, 4096] f32
  const float* W = (const float*)d_in[1];  // [128, 4096]   f32
  float* out = (float*)d_out;              // 131072 weights ++ 131072 indices

  const size_t wbBytes = (size_t)NEXP * HID * 2;  // 1 MB of bf16 W
  dim3 grid(NUM_TOKENS / TOK_BLK);                // 256 blocks
  dim3 block(256);                                // 8 waves (wave32)

  if (ws_size >= wbBytes) {
    __bf16* Wb = (__bf16*)d_ws;
    hipLaunchKernelGGL(w_to_bf16_kernel, dim3((NEXP * HID) / (256 * 8)), dim3(256),
                       0, stream, W, Wb);
    hipLaunchKernelGGL((moe_router_kernel<true>), grid, block, 0, stream,
                       x, W, (const __bf16*)Wb, out);
  } else {
    hipLaunchKernelGGL((moe_router_kernel<false>), grid, block, 0, stream,
                       x, W, (const __bf16*)nullptr, out);
  }
}
